// CGnet_30605936951620
// MI455X (gfx1250) — matrix operand order repaired and checked by reading, not compiled, not run on hardware
//
#include <hip/hip_runtime.h>
#include <math.h>

typedef __attribute__((ext_vector_type(16))) _Float16 v16h;
typedef __attribute__((ext_vector_type(8)))  float    v8f;

#define NATOMS 1024
#define KNEIGH 32
#define TCH    16
#define CG_TOTAL 3436

// ---- pair tables: for each output l, (l1,l2) with |l1-l2|<=l<=l1+l2, l1-major ----
constexpr int cNP[4] = {4, 9, 11, 10};
constexpr int cP1[4][11] = {
  {0,1,2,3, 0,0,0,0,0,0,0},
  {0,1,1,1,2,2,2,3,3, 0,0},
  {0,1,1,1,2,2,2,2,3,3,3},
  {0,1,1,2,2,2,3,3,3,3, 0}};
constexpr int cP2[4][11] = {
  {0,1,2,3, 0,0,0,0,0,0,0},
  {1,0,1,2,1,2,3,2,3, 0,0},
  {2,1,2,3,0,1,2,3,1,2,3},
  {3,2,3,1,2,3,0,1,2,3, 0}};
constexpr int cCGOFF[4][11] = {
  {0,1,10,35, 0,0,0,0,0,0,0},
  {84,93,102,129,174,219,294,399,504, 0,0},
  {651,676,721,796,901,926,1001,1126,1301,1406,1581},
  {1826,1875,1980,2127,2232,2407,2652,2701,2848,3093, 0}};
// complex feature layout: feat[l] at cFOFF[l] + t*(2l+1) + m  (cFOFF[l] = 16*l*l)
constexpr int cFOFF[4] = {0, 16, 64, 144};

// Non-convergent scheduling fence: keeps LDS store->load order without
// blocking loop unrolling (hardware LDS is in-order within a wave).
__device__ __forceinline__ void lds_fence() { asm volatile("" ::: "memory"); }

// ================= CG coefficient init kernel (device double math) ============
__device__ double lgf(int n) {
  double s = 0.0;
  for (int i = 2; i <= n; ++i) s += log((double)i);
  return s;
}

__device__ float cg_val(int j1, int m1, int j2, int m2, int J, int M) {
  if (m1 + m2 != M || J < abs(j1 - j2) || J > j1 + j2) return 0.0f;
  double pre = 0.5 * (log((double)(2 * J + 1)) + lgf(J + j1 - j2) + lgf(J - j1 + j2) +
                      lgf(j1 + j2 - J) - lgf(j1 + j2 + J + 1) + lgf(J + M) + lgf(J - M) +
                      lgf(j1 - m1) + lgf(j1 + m1) + lgf(j2 - m2) + lgf(j2 + m2));
  int kmin = max(0, max(j2 - J - m1, j1 + m2 - J));
  int kmax = min(j1 + j2 - J, min(j1 - m1, j2 + m2));
  double s = 0.0;
  for (int k = kmin; k <= kmax; ++k) {
    double t = exp(pre - (lgf(k) + lgf(j1 + j2 - J - k) + lgf(j1 - m1 - k) +
                          lgf(j2 + m2 - k) + lgf(J - j2 + m1 + k) + lgf(J - j1 - m2 + k)));
    s += (k & 1) ? -t : t;
  }
  return (float)s;
}

__global__ void cg_init_kernel(float* __restrict__ cg) {
  int tid = blockIdx.x * blockDim.x + threadIdx.x;
  int stride = gridDim.x * blockDim.x;
  for (int idx = tid; idx < CG_TOTAL; idx += stride) {
    int l = 0, p = 0, off = 0;
    for (int ll = 0; ll < 4; ++ll)
      for (int pp = 0; pp < cNP[ll]; ++pp) {
        int o  = cCGOFF[ll][pp];
        int sz = (2 * ll + 1) * (2 * cP1[ll][pp] + 1) * (2 * cP2[ll][pp] + 1);
        if (idx >= o && idx < o + sz) { l = ll; p = pp; off = o; }
      }
    int l1 = cP1[l][p], l2 = cP2[l][p];
    int d1 = 2 * l1 + 1, d2 = 2 * l2 + 1;
    int loc = idx - off;
    int M = loc / (d1 * d2);
    int a = (loc / d2) % d1;
    int b = loc % d2;
    cg[idx] = cg_val(l1, a - l1, l2, b - l2, l, M - l);
  }
}

__global__ void zero_out_kernel(float* o) {
  if (threadIdx.x == 0 && blockIdx.x == 0) o[0] = 0.0f;
}

// ================================ main kernel =================================
struct Params {
  const float* nl;
  const float* cw[3][4];
  const float* gamma;
  const float* beta;
  const float* mw[6];
  const float* mb[6];
  const float* cg;
  float* out;
};

__device__ __forceinline__ float ffact(int n) {
  float r = 1.0f;
  for (int i = 2; i <= n; ++i) r *= (float)i;
  return r;
}

// ---- conv layer 0: 1 input channel, K = npairs(l) <= 11: scalar path ----
template <int l>
__device__ __forceinline__ void conv_first(const float* __restrict__ W0,
                                           const float* __restrict__ gCG,
                                           const float2* __restrict__ fin,
                                           float2* __restrict__ fout,
                                           int lane) {
  constexpr int NPl = cNP[l];
  const int t = lane & 15;
  if ((lane >> 4) == 0) {
    #pragma unroll
    for (int m = 0; m < 2 * l + 1; ++m) {
      float ore = 0.0f, oim = 0.0f;
      #pragma unroll
      for (int p = 0; p < NPl; ++p) {
        const int l1 = cP1[l][p], l2 = cP2[l][p];
        const int d1 = 2 * l1 + 1, d2 = 2 * l2 + 1;
        float sre = 0.0f, sim = 0.0f;
        #pragma unroll
        for (int a = 0; a < 7; ++a) {
          if (a < d1) {
            const int b = (m - l) - (a - l1) + l2;
            if (b >= 0 && b < d2) {  // fully compile-time
              float cgc = gCG[cCGOFF[l][p] + (m * d1 + a) * d2 + b];  // uniform
              float2 u = fin[cFOFF[l1] + a];
              float2 q = fin[cFOFF[l2] + b];
              sre += cgc * (u.x * q.x - u.y * q.y);
              sim += cgc * (u.x * q.y + u.y * q.x);
            }
          }
        }
        const float wv = W0[t * NPl + p];
        ore += wv * sre;
        oim += wv * sim;
      }
      fout[cFOFF[l] + t * (2 * l + 1) + m] = make_float2(ore, oim);
    }
  }
}

// ---- one (l, pair) slab of a dense conv: 8 blocks of K=32, WMMA-accumulated ----
template <int l, int pair>
__device__ __forceinline__ void conv_pair(const float* __restrict__ Wt,
                                          const float* __restrict__ gCG,
                                          const float2* __restrict__ fin,
                                          float* __restrict__ stage,
                                          int lane, int c, int h, int cc,
                                          v8f& acc) {
  constexpr int D    = 2 * l + 1;
  constexpr int l1   = cP1[l][pair];
  constexpr int l2   = cP2[l][pair];
  constexpr int d1   = 2 * l1 + 1;
  constexpr int d2   = 2 * l2 + 1;
  constexpr int off  = l - l1 - l2;
  constexpr int Ktot = cNP[l] * 256;
  const float* __restrict__ cgp = gCG + cCGOFF[l][pair];

  // CG coefficients: loop-invariant; pull into registers BEFORE any fence so
  // the asm memory clobber cannot force per-iteration reloads.
  float cgv[d1 * d2];
  #pragma unroll
  for (int a = 0; a < d1; ++a) {
    #pragma unroll
    for (int b = 0; b < d2; ++b) {
      const int M = a + b + off;
      if (M >= 0 && M < D) cgv[a * d2 + b] = cgp[(M * d1 + a) * d2 + b];
    }
  }

  // x2 row: depends only on lane column j=c -> invariant across sb, hoist.
  const float2* __restrict__ x2 = &fin[cFOFF[l2] + c * d2];
  float2 q[d2];
  #pragma unroll
  for (int b = 0; b < d2; ++b) q[b] = x2[b];

  for (int sb = 0; sb < 8; ++sb) {
    float* __restrict__ st = stage + (sb & 1) * (14 * 32);
    // -------- phase 1: cooperative CG expansion, lane = K-row --------
    const int i = 2 * sb + h;  // x[l1] channel
    const float2* __restrict__ x1 = &fin[cFOFF[l1] + i * d1];
    float2 u[d1];
    #pragma unroll
    for (int a = 0; a < d1; ++a) u[a] = x1[a];
    float ore[D], oim[D];
    #pragma unroll
    for (int m = 0; m < D; ++m) { ore[m] = 0.0f; oim[m] = 0.0f; }
    #pragma unroll
    for (int a = 0; a < d1; ++a) {
      #pragma unroll
      for (int b = 0; b < d2; ++b) {
        const int M = a + b + off;          // compile-time after unroll
        if (M >= 0 && M < D) {
          const float cgc = cgv[a * d2 + b];
          ore[M] += cgc * (u[a].x * q[b].x - u[a].y * q[b].y);
          oim[M] += cgc * (u[a].x * q[b].y + u[a].y * q[b].x);
        }
      }
    }
    #pragma unroll
    for (int m = 0; m < D; ++m) {
      st[(2 * m + 0) * 32 + lane] = ore[m];
      st[(2 * m + 1) * 32 + lane] = oim[m];
    }
    lds_fence();
    // -------- phase 2: fragments + WMMA --------
    const int kbase = pair * 256 + sb * 32;
    const float* wrow = Wt + c * Ktot + kbase + 8 * h;
    const float4 a0 = *(const float4*)(wrow + 0);
    const float4 a1 = *(const float4*)(wrow + 4);
    const float4 a2 = *(const float4*)(wrow + 16);
    const float4 a3 = *(const float4*)(wrow + 20);
    const float* bcol = st + cc * 32 + 8 * h;  // B: e<8 -> k = 8h+e, e>=8 -> k = 16+8h+(e-8)
    const float4 b0 = *(const float4*)(bcol + 0);
    const float4 b1 = *(const float4*)(bcol + 4);
    const float4 b2 = *(const float4*)(bcol + 16);
    const float4 b3 = *(const float4*)(bcol + 20);
    v16h afrag, bfrag;
    afrag[0]  = (_Float16)a0.x; afrag[1]  = (_Float16)a0.y;
    afrag[2]  = (_Float16)a0.z; afrag[3]  = (_Float16)a0.w;
    afrag[4]  = (_Float16)a1.x; afrag[5]  = (_Float16)a1.y;
    afrag[6]  = (_Float16)a1.z; afrag[7]  = (_Float16)a1.w;
    afrag[8]  = (_Float16)a2.x; afrag[9]  = (_Float16)a2.y;
    afrag[10] = (_Float16)a2.z; afrag[11] = (_Float16)a2.w;
    afrag[12] = (_Float16)a3.x; afrag[13] = (_Float16)a3.y;
    afrag[14] = (_Float16)a3.z; afrag[15] = (_Float16)a3.w;
    bfrag[0]  = (_Float16)b0.x; bfrag[1]  = (_Float16)b0.y;
    bfrag[2]  = (_Float16)b0.z; bfrag[3]  = (_Float16)b0.w;
    bfrag[4]  = (_Float16)b1.x; bfrag[5]  = (_Float16)b1.y;
    bfrag[6]  = (_Float16)b1.z; bfrag[7]  = (_Float16)b1.w;
    bfrag[8]  = (_Float16)b2.x; bfrag[9]  = (_Float16)b2.y;
    bfrag[10] = (_Float16)b2.z; bfrag[11] = (_Float16)b2.w;
    bfrag[12] = (_Float16)b3.x; bfrag[13] = (_Float16)b3.y;
    bfrag[14] = (_Float16)b3.z; bfrag[15] = (_Float16)b3.w;
    acc = __builtin_amdgcn_wmma_f32_16x16x32_f16(false, afrag, false, bfrag,
                                                 (short)0, acc, false, false);
    // no second fence: double-buffered staging + in-order LDS cover the WAR
    // hazard; conflicting reuse is 2 blocks away, past the next block's fence.
  }
}

template <int l, int pair>
__device__ __forceinline__ void conv_pairs(const float* __restrict__ Wt,
                                           const float* __restrict__ gCG,
                                           const float2* __restrict__ fin,
                                           float* __restrict__ stage,
                                           int lane, int c, int h, int cc,
                                           v8f& acc) {
  conv_pair<l, pair>(Wt, gCG, fin, stage, lane, c, h, cc, acc);
  if constexpr (pair + 1 < cNP[l])
    conv_pairs<l, pair + 1>(Wt, gCG, fin, stage, lane, c, h, cc, acc);
}

// ---- conv layers 1..2: 16 input channels, K = 256*npairs(l), WMMA path ----
template <int l>
__device__ __forceinline__ void conv_dense(const float* __restrict__ Wt,
                                           const float* __restrict__ gCG,
                                           const float2* __restrict__ fin,
                                           float2* __restrict__ fout,
                                           float* __restrict__ stage,  // [2][14*32]
                                           int lane) {
  constexpr int D     = 2 * l + 1;
  constexpr int usedN = 2 * D;
  const int c = lane & 15, h = lane >> 4;
  const int cc = (c < usedN) ? c : (usedN - 1);  // clamped col (garbage unused)
  v8f acc = {};
  conv_pairs<l, 0>(Wt, gCG, fin, stage, lane, c, h, cc, acc);
  // D: lane col c, VGPR r -> row r+8h
  if (c < usedN) {
    const int m = c >> 1;
    #pragma unroll
    for (int r = 0; r < 8; ++r) {
      const int t = r + 8 * h;
      const int idx = cFOFF[l] + t * D + m;
      if (c & 1) fout[idx].y = acc[r];
      else       fout[idx].x = acc[r];
    }
  }
}

__global__ __launch_bounds__(128) void cgnet_kernel(Params P) {
  __shared__ __align__(16) float2 sFeat[2][256];
  __shared__ __align__(16) float  sStage[4][2 * 14 * 32];
  __shared__ float sV[48];
  __shared__ float sMLP[2][64];
  __shared__ float s_mu, s_rstd;

  const int tid  = threadIdx.x;
  const int w    = __builtin_amdgcn_readfirstlane(tid >> 5);  // uniform wave id
  const int lane = tid & 31;
  const int atom = blockIdx.x;

  // ---- spherical harmonics: wave 0, one neighbor per lane ----
  if (w == 0) {
    const float* pos = P.nl + (size_t)(atom * KNEIGH + lane) * 3;
    float x = pos[0], y = pos[1], z = pos[2];
    float r   = sqrtf(x * x + y * y + z * z + 1e-12f);
    float ct  = z / r;
    float phi = atan2f(y, x);
    float st  = sqrtf(fmaxf(1.0f - ct * ct, 1e-12f));
    float Pl[4][4];
    Pl[0][0] = 1.0f;
    Pl[1][1] = Pl[0][0] * -1.0f * st;
    Pl[2][2] = Pl[1][1] * -3.0f * st;
    Pl[3][3] = Pl[2][2] * -5.0f * st;
    Pl[1][0] = 1.0f * ct * Pl[0][0];
    Pl[2][1] = 3.0f * ct * Pl[1][1];
    Pl[3][2] = 5.0f * ct * Pl[2][2];
    #pragma unroll
    for (int m = 0; m <= 3; ++m)
      #pragma unroll
      for (int l = m + 2; l <= 3; ++l)
        Pl[l][m] = ((2 * l - 1) * ct * Pl[l - 1][m] - (l + m - 1) * Pl[l - 2][m]) / (float)(l - m);
    float fc = (r < 5.0f) ? 0.5f * (cosf(3.14159265358979323846f * r / 5.0f) + 1.0f) : 0.0f;

    float yre[16], yim[16];
    #pragma unroll
    for (int l = 0; l <= 3; ++l) {
      #pragma unroll
      for (int m = 0; m <= l; ++m) {
        float Klm = sqrtf((2.0f * l + 1.0f) / (4.0f * 3.14159265358979323846f) *
                          ffact(l - m) / ffact(l + m));
        float mag = Klm * Pl[l][m] * fc;
        float cre = mag * cosf(m * phi);
        float cim = mag * sinf(m * phi);
        yre[l * l + (l + m)] = cre;
        yim[l * l + (l + m)] = cim;
        if (m > 0) {
          float sgn = (m & 1) ? -1.0f : 1.0f;
          yre[l * l + (l - m)] = sgn * cre;
          yim[l * l + (l - m)] = -sgn * cim;
        }
      }
    }
    #pragma unroll
    for (int e = 0; e < 16; ++e) {
      for (int off = 16; off >= 1; off >>= 1) {
        yre[e] += __shfl_xor(yre[e], off, 32);
        yim[e] += __shfl_xor(yim[e], off, 32);
      }
    }
    if (lane == 0) {
      #pragma unroll
      for (int e = 0; e < 16; ++e) sFeat[0][e] = make_float2(yre[e], yim[e]);
    }
  }
  __syncthreads();

  // ---- 3 CG-conv layers, wave w owns output order l=w ----
  int cur = 0;
  for (int n = 0; n < 3; ++n) {
    const float2* fin = &sFeat[cur][0];
    float2* fout = &sFeat[1 - cur][0];
    float* stg = &sStage[w][0];
    if (n == 0) {
      switch (w) {
        case 0:  conv_first<0>(P.cw[0][0], P.cg, fin, fout, lane); break;
        case 1:  conv_first<1>(P.cw[0][1], P.cg, fin, fout, lane); break;
        case 2:  conv_first<2>(P.cw[0][2], P.cg, fin, fout, lane); break;
        default: conv_first<3>(P.cw[0][3], P.cg, fin, fout, lane); break;
      }
    } else {
      switch (w) {
        case 0:  conv_dense<0>(P.cw[n][0], P.cg, fin, fout, stg, lane); break;
        case 1:  conv_dense<1>(P.cw[n][1], P.cg, fin, fout, stg, lane); break;
        case 2:  conv_dense<2>(P.cw[n][2], P.cg, fin, fout, stg, lane); break;
        default: conv_dense<3>(P.cw[n][3], P.cg, fin, fout, stg, lane); break;
      }
    }
    __syncthreads();
    if (tid < TCH) sV[n * TCH + tid] = fout[tid].x;  // real(x[0][:,0])
    cur ^= 1;
  }
  __syncthreads();

  // ---- LayerNorm + MLP ----
  if (tid == 0) {
    float mu = 0.0f;
    for (int i = 0; i < 48; ++i) mu += sV[i];
    mu /= 48.0f;
    float var = 0.0f;
    for (int i = 0; i < 48; ++i) { float d = sV[i] - mu; var += d * d; }
    var /= 48.0f;
    s_mu = mu;
    s_rstd = rsqrtf(var + 1e-5f);
  }
  __syncthreads();
  if (tid < 48)
    sMLP[0][tid] = (sV[tid] - s_mu) * s_rstd * P.gamma[tid] + P.beta[tid];
  __syncthreads();

  const int dims[7] = {48, 64, 64, 64, 64, 64, 1};
  int buf = 0;
  for (int layer = 0; layer < 5; ++layer) {
    const int din = dims[layer];
    if (tid < 64) {
      const float* __restrict__ Wm = P.mw[layer];
      float s = P.mb[layer][tid];
      for (int k2 = 0; k2 < din; ++k2) s += Wm[tid * din + k2] * sMLP[buf][k2];
      sMLP[1 - buf][tid] = tanhf(s);
    }
    __syncthreads();
    buf = 1 - buf;
  }
  if (tid == 0) {
    float s = P.mb[5][0];
    const float* __restrict__ Wm = P.mw[5];
    for (int k2 = 0; k2 < 64; ++k2) s += Wm[k2] * sMLP[buf][k2];
    atomicAdd(P.out, s);
  }
}

// ================================= launcher ==================================
extern "C" void kernel_launch(void* const* d_in, const int* in_sizes, int n_in,
                              void* d_out, int out_size, void* d_ws, size_t ws_size,
                              hipStream_t stream) {
  (void)in_sizes; (void)n_in; (void)out_size; (void)ws_size;
  Params p;
  p.nl = (const float*)d_in[0];
  for (int n = 0; n < 3; ++n)
    for (int l = 0; l < 4; ++l)
      p.cw[n][l] = (const float*)d_in[1 + n * 4 + l];
  p.gamma = (const float*)d_in[13];
  p.beta  = (const float*)d_in[14];
  for (int i = 0; i < 6; ++i) {
    p.mw[i] = (const float*)d_in[15 + 2 * i];
    p.mb[i] = (const float*)d_in[16 + 2 * i];
  }
  p.cg  = (const float*)d_ws;
  p.out = (float*)d_out;

  zero_out_kernel<<<1, 64, 0, stream>>>(p.out);
  cg_init_kernel<<<14, 256, 0, stream>>>((float*)d_ws);
  cgnet_kernel<<<NATOMS, 128, 0, stream>>>(p);
}
